// SingleStreamMutiAttention_16484084483153
// MI455X (gfx1250) — matrix operand order repaired and checked by reading, not compiled, not run on hardware
//
#include <hip/hip_runtime.h>

#define NT      21
#define S_LEN   1560
#define C_DIM   2048
#define H_HEADS 16
#define HD      128
#define NA      32
#define E_DIM   768
#define L_TOK   (NT * S_LEN)          // 32760
#define ATT_SCALE 0.08838834764831845f  // 1/sqrt(128)
#define LOG2_1E4_OVER64 (13.287712379549449f / 64.0f)
#define EPSV 1e-8f

typedef _Float16 v16h __attribute__((ext_vector_type(16)));
typedef _Float16 v8h  __attribute__((ext_vector_type(8)));
typedef _Float16 v4h  __attribute__((ext_vector_type(4)));
typedef float    v8f  __attribute__((ext_vector_type(8)));
typedef unsigned int u32x4 __attribute__((ext_vector_type(4)));
typedef int      i32x8 __attribute__((ext_vector_type(8)));

// ---------------------------------------------------------------------------
// Cross-lane XOR exchange via ds_swizzle (group-of-32 mode, imm pattern)
// offset = {0, xor_mask[14:10], or_mask[9:5]=0, and_mask[4:0]=0x1F}
// ---------------------------------------------------------------------------
#define XOR_SWZ(x, mask) \
  __int_as_float(__builtin_amdgcn_ds_swizzle(__float_as_int(x), (0x1F | ((mask) << 10))))

__device__ __forceinline__ float grp16_max(float x) {
  x = fmaxf(x, XOR_SWZ(x, 1));
  x = fmaxf(x, XOR_SWZ(x, 2));
  x = fmaxf(x, XOR_SWZ(x, 4));
  x = fmaxf(x, XOR_SWZ(x, 8));
  return x;
}
__device__ __forceinline__ float grp16_sum(float x) {
  x += XOR_SWZ(x, 1);
  x += XOR_SWZ(x, 2);
  x += XOR_SWZ(x, 4);
  x += XOR_SWZ(x, 8);
  return x;
}

// ---------------------------------------------------------------------------
// Fragment loaders (16-bit WMMA layouts per CDNA5 ISA 7.12.2)
// A 16x32 frag: lane m=l%16, g=l/16; VGPR0-3 hold K=g*8+0..7, VGPR4-7 K=16+g*8+0..7
// B 32x16 frag: lane n=l%16, g=l/16; VGPR0-3 hold K=g*8+0..7, VGPR4-7 K=16+g*8+0..7 of column n
// ---------------------------------------------------------------------------
__device__ __forceinline__ v16h h_cat(v8h lo, v8h hi) {
  return __builtin_shufflevector(lo, hi, 0,1,2,3,4,5,6,7,8,9,10,11,12,13,14,15);
}

__device__ __forceinline__ v16h ld_afrag(const _Float16* sm, int stride, int rowBase,
                                         int kOff, int lane) {
  int m = lane & 15, g = lane >> 4;
  const _Float16* p = sm + (rowBase + m) * stride + kOff + g * 8;
  v8h lo = *(const v8h*)p;
  v8h hi = *(const v8h*)(p + 16);
  return h_cat(lo, hi);
}

__device__ __forceinline__ v16h ld_bfrag(const _Float16* sm, int stride, int colBase,
                                         int kOff, int lane) {
  int n = lane & 15, g = lane >> 4;
  const _Float16* p = sm + (colBase + n) * stride + kOff + g * 8;
  v8h lo = *(const v8h*)p;
  v8h hi = *(const v8h*)(p + 16);
  return h_cat(lo, hi);
}

// ---------------------------------------------------------------------------
// LDS staging: 128 rows x 32 cols slab, f32 -> f16 convert path
// ---------------------------------------------------------------------------
__device__ __forceinline__ void ld4cvt(const float* s, v4h& d) {
  float4 v = *(const float4*)s;
  d[0] = (_Float16)v.x; d[1] = (_Float16)v.y;
  d[2] = (_Float16)v.z; d[3] = (_Float16)v.w;
}
__device__ __forceinline__ void ld4cvt(const _Float16* s, v4h& d) {
  d = *(const v4h*)s;
}

template <typename AT>
__device__ __forceinline__ void stage_tile(const AT* __restrict__ G, long long rowBase,
                                           long long rowMax, long long ld, int k0,
                                           int Ktot, _Float16* sm, int tid) {
#pragma unroll
  for (int i = 0; i < 4; i++) {
    int idx = tid + i * 256;
    int row = idx >> 3;
    int c4  = (idx & 7) << 2;
    long long gr = rowBase + row;
    if (gr > rowMax) gr = rowMax;
    const AT* src = G + gr * ld + k0 + c4;
    if (k0 + 32 < Ktot) __builtin_prefetch((const void*)(src + 32), 0, 1);
    v4h h;
    ld4cvt(src, h);
    *(v4h*)(sm + row * 40 + c4) = h;
  }
}

// f16 A staging via CDNA5 async global->LDS copy (ASYNCcnt-tracked, no VGPR data)
__device__ __forceinline__ void stage_tile_async_f16(const _Float16* __restrict__ G,
                                                     long long rowBase, long long rowMax,
                                                     long long ld, int k0,
                                                     _Float16* sm, int tid) {
#pragma unroll
  for (int i = 0; i < 2; i++) {
    int idx = tid + i * 256;
    int row = idx >> 2;
    int c16 = (idx & 3) << 3;          // half-offset: 0,8,16,24 (16B chunks)
    long long gr = rowBase + row;
    if (gr > rowMax) gr = rowMax;
    const _Float16* src = G + gr * ld + k0 + c16;
    unsigned lds_off = (unsigned)(unsigned long long)(sm + row * 40 + c16);
    asm volatile("global_load_async_to_lds_b128 %0, %1, off"
                 :: "v"(lds_off), "v"(src) : "memory");
  }
}

// ---------------------------------------------------------------------------
// Tiled WMMA GEMM: Out[M,N] = A[M,K] * Bw[N,K]^T + bias[N]
// block = 256 threads (8 waves), tile 128x128, BK=32, double-buffered LDS
// ---------------------------------------------------------------------------
template <typename AT, typename OT>
__global__ void __launch_bounds__(256) gemm_wmma_kernel(
    const AT* __restrict__ A, const float* __restrict__ Bw,
    const float* __restrict__ bias, OT* __restrict__ Out, int M, int N, int K) {
  __shared__ _Float16 smA[2][128 * 40];
  __shared__ _Float16 smB[2][128 * 40];
  const int tid  = threadIdx.x;
  const int lane = tid & 31;
  const int w    = tid >> 5;
  const int wm   = w & 1;    // 2 wave-rows of 64
  const int wn   = w >> 1;   // 4 wave-cols of 32
  const long long rowBase = (long long)blockIdx.y * 128;
  const long long colBase = (long long)blockIdx.x * 128;

  v8f zero = {};
  v8f acc[4][2];
#pragma unroll
  for (int i = 0; i < 4; i++)
#pragma unroll
    for (int j = 0; j < 2; j++) acc[i][j] = zero;

  const int KT = K >> 5;
  if constexpr (__is_same(AT, _Float16)) {
    stage_tile_async_f16(A, rowBase, (long long)M - 1, K, 0, smA[0], tid);
  } else {
    stage_tile(A, rowBase, (long long)M - 1, K, 0, K, smA[0], tid);
  }
  stage_tile(Bw, colBase, (long long)N - 1, K, 0, K, smB[0], tid);
  if constexpr (__is_same(AT, _Float16)) {
    asm volatile("s_wait_asynccnt 0x0" ::: "memory");
  }
  __syncthreads();

  for (int kt = 0; kt < KT; ++kt) {
    const int cur = kt & 1;
    if (kt + 1 < KT) {
      if constexpr (__is_same(AT, _Float16)) {
        stage_tile_async_f16(A, rowBase, (long long)M - 1, K, (kt + 1) << 5, smA[cur ^ 1], tid);
      } else {
        stage_tile(A, rowBase, (long long)M - 1, K, (kt + 1) << 5, K, smA[cur ^ 1], tid);
      }
      stage_tile(Bw, colBase, (long long)N - 1, K, (kt + 1) << 5, K, smB[cur ^ 1], tid);
    }
    v16h af[4], bf[2];
#pragma unroll
    for (int i = 0; i < 4; i++) af[i] = ld_afrag(smA[cur], 40, wm * 64 + i * 16, 0, lane);
#pragma unroll
    for (int j = 0; j < 2; j++) bf[j] = ld_bfrag(smB[cur], 40, wn * 32 + j * 16, 0, lane);
#pragma unroll
    for (int i = 0; i < 4; i++)
#pragma unroll
      for (int j = 0; j < 2; j++)
        acc[i][j] = __builtin_amdgcn_wmma_f32_16x16x32_f16(
            false, af[i], false, bf[j], (short)0, acc[i][j], false, false);
    if constexpr (__is_same(AT, _Float16)) {
      asm volatile("s_wait_asynccnt 0x0" ::: "memory");
    }
    __syncthreads();
  }

#pragma unroll
  for (int j = 0; j < 2; j++) {
    long long col = colBase + wn * 32 + j * 16 + (lane & 15);
    float bv = bias[col];
#pragma unroll
    for (int i = 0; i < 4; i++) {
#pragma unroll
      for (int r = 0; r < 8; r++) {
        long long row = rowBase + wm * 64 + i * 16 + (lane >> 4) * 8 + r;
        if (row < M) Out[row * (long long)N + col] = (OT)(acc[i][j][r] + bv);
      }
    }
  }
}

// ---------------------------------------------------------------------------
// Row min/max of x_ref_attn_map (2 x L) -> stats[4]
// ---------------------------------------------------------------------------
__global__ void __launch_bounds__(256) stats_kernel(const float* __restrict__ amap,
                                                    int L, float* __restrict__ stats) {
  __shared__ float r0n[256], r0x[256], r1n[256], r1x[256];
  int t = threadIdx.x;
  float mn0 = INFINITY, mx0 = -INFINITY, mn1 = INFINITY, mx1 = -INFINITY;
  for (int i = t; i < L; i += 256) {
    float a = amap[i], b = amap[L + i];
    mn0 = fminf(mn0, a); mx0 = fmaxf(mx0, a);
    mn1 = fminf(mn1, b); mx1 = fmaxf(mx1, b);
  }
  r0n[t] = mn0; r0x[t] = mx0; r1n[t] = mn1; r1x[t] = mx1;
  __syncthreads();
  for (int off = 128; off > 0; off >>= 1) {
    if (t < off) {
      r0n[t] = fminf(r0n[t], r0n[t + off]); r0x[t] = fmaxf(r0x[t], r0x[t + off]);
      r1n[t] = fminf(r1n[t], r1n[t + off]); r1x[t] = fmaxf(r1x[t], r1x[t + off]);
    }
    __syncthreads();
  }
  if (t == 0) { stats[0] = r0n[0]; stats[1] = r0x[0]; stats[2] = r1n[0]; stats[3] = r1x[0]; }
}

__global__ void __launch_bounds__(256) pos_kernel(const float* __restrict__ amap,
                                                  const float* __restrict__ stats,
                                                  float* __restrict__ pos, int L) {
  int i = blockIdx.x * 256 + threadIdx.x;
  if (i >= L) return;
  float m0 = amap[i], m1 = amap[L + i];
  float h1 = (m0 - stats[0]) / (stats[1] - stats[0] + EPSV) * 4.0f;         // [0,4]
  float h2 = (m1 - stats[2]) / (stats[3] - stats[2] + EPSV) * 4.0f + 20.0f; // [20,24]
  pos[i] = (m0 >= m1) ? h1 : h2;
}

// ---------------------------------------------------------------------------
// In-place RoPE on Q16 (interleaved pairs), per-token position
// ---------------------------------------------------------------------------
__global__ void __launch_bounds__(256) rope_q_kernel(_Float16* __restrict__ Q16,
                                                     const float* __restrict__ pos) {
  long long i = blockIdx.x;                     // token
  int hj = blockIdx.y * 256 + threadIdx.x;      // 0..1023 = head*64 + j
  int h = hj >> 6, j = hj & 63;
  float p = pos[i];
  float ang = p * exp2f(-(float)j * LOG2_1E4_OVER64);
  float sn, cs;
  __sincosf(ang, &sn, &cs);
  long long base = i * C_DIM + h * HD + 2 * j;
  float a = (float)Q16[base], b = (float)Q16[base + 1];
  Q16[base]     = (_Float16)(a * cs - b * sn);
  Q16[base + 1] = (_Float16)(b * cs + a * sn);
}

// ---------------------------------------------------------------------------
// Attention: per (frame, head, 128-row S tile). Na=32 keys, hd=128.
// Q tile staged via Tensor Data Mover; scores & P*V via WMMA;
// softmax via ds_swizzle over 16-lane groups.
// ---------------------------------------------------------------------------
__global__ void __launch_bounds__(256) attn_kernel(const _Float16* __restrict__ Q16,
                                                   const _Float16* __restrict__ KV16,
                                                   _Float16* __restrict__ O16) {
  __shared__ _Float16 smQ[128 * 136];
  __shared__ _Float16 smK[32 * 136];
  __shared__ _Float16 smVt[128 * 40];
  __shared__ _Float16 smP[8 * 16 * 40];

  const int tid  = threadIdx.x;
  const int lane = tid & 31;
  const int w    = tid >> 5;        // wave: 16 S-rows each
  const int t    = blockIdx.x >> 4;
  const int h    = blockIdx.x & 15;
  const int s0   = blockIdx.y * 128;

  // --- stage Q tile (128 rows x 128 halves, LDS stride 136) via TDM -------
  // D# group0: count=1, lds_addr, global_addr, type=2
  // D# group1: data_size=2B, pad_enable, pad_interval=64 DW (256B),
  //            pad_amount=4 DW (16B) -> LDS row stride 272B = 136 halves;
  //            tile 128x128, tensor_dim0=128, tensor_dim1=S-s0 (OOB rows -> 0),
  //            tensor_dim0_stride=2048
  if (tid == 0) {
    unsigned long long gaddr =
        (unsigned long long)(Q16 + ((long long)(t * S_LEN + s0)) * C_DIM + h * HD);
    unsigned lds_addr = (unsigned)(unsigned long long)(&smQ[0]);
    unsigned rows_rem = (unsigned)(S_LEN - s0);
    u32x4 g0;
    g0[0] = 1u;                                       // count=1, user mode
    g0[1] = lds_addr;
    g0[2] = (unsigned)(gaddr & 0xFFFFFFFFull);
    g0[3] = (unsigned)((gaddr >> 32) & 0x01FFFFFFull) | (2u << 30);  // type=2
    i32x8 g1;
    g1[0] = (1 << 16) | (1 << 20) | (5 << 22) | (3 << 25);
    g1[1] = (int)((128u & 0xFFFFu) << 16);            // tensor_dim0[15:0]=128
    g1[2] = (int)((rows_rem & 0xFFFFu) << 16);        // tensor_dim1[15:0]
    g1[3] = (int)((rows_rem >> 16) & 0xFFFFu) | (128 << 16);  // dim1 hi | tile_dim0=128
    g1[4] = 128;                                      // tile_dim1=128 rows
    g1[5] = C_DIM;                                    // tensor_dim0_stride
    g1[6] = 0;
    g1[7] = 0;
    asm volatile("tensor_load_to_lds %0, %1" :: "s"(g0), "s"(g1) : "memory");
    __builtin_amdgcn_s_wait_tensorcnt(0);
  }

  // stage K with RoPE (enc_pos: a<16 -> 2.0, else 22.0)
#pragma unroll
  for (int i = 0; i < 8; i++) {
    int idx = tid + i * 256;
    int a = idx >> 6;
    int j = idx & 63;
    float p = (a < NA / 2) ? 2.0f : 22.0f;
    float ang = p * exp2f(-(float)j * LOG2_1E4_OVER64);
    float sn, cs;
    __sincosf(ang, &sn, &cs);
    long long base = ((long long)(t * NA + a)) * 4096 + h * HD + 2 * j;
    float k0v = (float)KV16[base];
    float k1v = (float)KV16[base + 1];
    smK[a * 136 + 2 * j]     = (_Float16)(k0v * cs - k1v * sn);
    smK[a * 136 + 2 * j + 1] = (_Float16)(k1v * cs + k0v * sn);
  }
  // stage V transposed: Vt[d][a]
#pragma unroll
  for (int i = 0; i < 16; i++) {
    int idx = tid + i * 256;
    int a = idx >> 7;
    int d = idx & 127;
    smVt[d * 40 + a] = KV16[((long long)(t * NA + a)) * 4096 + 2048 + h * HD + d];
  }
  __syncthreads();

  // scores = Q * K^T  (two 16-col key groups)
  v8f sc0 = {}, sc1 = {};
#pragma unroll
  for (int kc = 0; kc < 4; ++kc) {
    v16h qa  = ld_afrag(smQ, 136, w * 16, kc * 32, lane);
    v16h kb0 = ld_bfrag(smK, 136, 0,  kc * 32, lane);
    v16h kb1 = ld_bfrag(smK, 136, 16, kc * 32, lane);
    sc0 = __builtin_amdgcn_wmma_f32_16x16x32_f16(false, qa, false, kb0, (short)0, sc0, false, false);
    sc1 = __builtin_amdgcn_wmma_f32_16x16x32_f16(false, qa, false, kb1, (short)0, sc1, false, false);
  }

  // row softmax over 32 keys; C layout: lane holds col n=lane%16, row r+(lane/16)*8
#pragma unroll
  for (int r = 0; r < 8; r++) {
    float x0 = sc0[r] * ATT_SCALE;
    float x1 = sc1[r] * ATT_SCALE;
    float mx  = grp16_max(fmaxf(x0, x1));
    float e0 = __expf(x0 - mx);
    float e1 = __expf(x1 - mx);
    float sum = grp16_sum(e0 + e1);
    float inv = 1.0f / sum;
    int row = (lane >> 4) * 8 + r;
    int n = lane & 15;
    smP[(w * 16 + row) * 40 + n]      = (_Float16)(e0 * inv);
    smP[(w * 16 + row) * 40 + 16 + n] = (_Float16)(e1 * inv);
  }
  __syncthreads();

  // O = P (16x32) * V (32x128) -> 8 N-fragments
  v16h pa = ld_afrag(smP, 40, w * 16, 0, lane);
#pragma unroll
  for (int nf = 0; nf < 8; ++nf) {
    v8f o = {};
    v16h vb = ld_bfrag(smVt, 40, nf * 16, 0, lane);
    o = __builtin_amdgcn_wmma_f32_16x16x32_f16(false, pa, false, vb, (short)0, o, false, false);
#pragma unroll
    for (int r = 0; r < 8; r++) {
      int sr = s0 + w * 16 + (lane >> 4) * 8 + r;
      if (sr < S_LEN)
        O16[((long long)(t * S_LEN + sr)) * C_DIM + h * HD + nf * 16 + (lane & 15)] =
            (_Float16)o[r];
    }
  }
}

// ---------------------------------------------------------------------------
extern "C" void kernel_launch(void* const* d_in, const int* in_sizes, int n_in,
                              void* d_out, int out_size, void* d_ws, size_t ws_size,
                              hipStream_t stream) {
  (void)in_sizes; (void)n_in; (void)out_size; (void)ws_size;
  const float* x    = (const float*)d_in[0];
  const float* enc  = (const float*)d_in[1];
  const float* amap = (const float*)d_in[2];
  const float* q_w  = (const float*)d_in[3];
  const float* q_b  = (const float*)d_in[4];
  const float* kv_w = (const float*)d_in[5];
  const float* kv_b = (const float*)d_in[6];
  const float* p_w  = (const float*)d_in[7];
  const float* p_b  = (const float*)d_in[8];
  float* out = (float*)d_out;

  char* wsp = (char*)d_ws;
  const size_t szQ  = (size_t)L_TOK * C_DIM * sizeof(_Float16);    // 134,184,960
  const size_t szKV = (size_t)(NT * NA) * 4096 * sizeof(_Float16); // 5,505,024
  _Float16* Q16  = (_Float16*)(wsp);
  _Float16* O16  = (_Float16*)(wsp + szQ);
  _Float16* KV16 = (_Float16*)(wsp + 2 * szQ);
  float* posb    = (float*)(wsp + 2 * szQ + szKV);
  float* stats   = (float*)(wsp + 2 * szQ + szKV + 131072);

  stats_kernel<<<1, 256, 0, stream>>>(amap, L_TOK, stats);
  pos_kernel<<<(L_TOK + 255) / 256, 256, 0, stream>>>(amap, stats, posb, L_TOK);

  // Q = x @ q_w^T + q_b   (M=32760, N=2048, K=2048)
  gemm_wmma_kernel<float, _Float16><<<dim3(C_DIM / 128, (L_TOK + 127) / 128), 256, 0, stream>>>(
      x, q_w, q_b, Q16, L_TOK, C_DIM, C_DIM);
  // KV = enc @ kv_w^T + kv_b (M=672, N=4096, K=768)
  gemm_wmma_kernel<float, _Float16><<<dim3(4096 / 128, (NT * NA + 127) / 128), 256, 0, stream>>>(
      enc, kv_w, kv_b, KV16, NT * NA, 4096, E_DIM);

  rope_q_kernel<<<dim3(L_TOK, 4), 256, 0, stream>>>(Q16, posb);

  attn_kernel<<<dim3(NT * H_HEADS, (S_LEN + 127) / 128), 256, 0, stream>>>(Q16, KV16, O16);

  // out = O @ proj_w^T + proj_b (f16 A via async-LDS staging, f32 out)
  gemm_wmma_kernel<_Float16, float><<<dim3(C_DIM / 128, (L_TOK + 127) / 128), 256, 0, stream>>>(
      O16, p_w, p_b, out, L_TOK, C_DIM, C_DIM);
}
